// Attention_50079318671492
// MI455X (gfx1250) — matrix operand order repaired and checked by reading, not compiled, not run on hardware
//
#include <hip/hip_runtime.h>
#include <hip/hip_bf16.h>
#include <math.h>

// Problem constants (from reference setup_inputs)
#define BB   4
#define HH   8
#define SS   8192
#define DD   64
#define SEG  1024
#define NSEG 8
#define QTILE 128
#define KTILE 64

typedef __attribute__((ext_vector_type(16))) _Float16 v16h;
typedef __attribute__((ext_vector_type(8)))  _Float16 v8h;
typedef __attribute__((ext_vector_type(4)))  _Float16 v4h;
typedef __attribute__((ext_vector_type(2)))  _Float16 v2h;
typedef __attribute__((ext_vector_type(8)))  float    v8f;
typedef __attribute__((ext_vector_type(4)))  float    f32x4;

__device__ __forceinline__ v16h cat8(v8h a, v8h b) {
    return __builtin_shufflevector(a, b, 0,1,2,3,4,5,6,7,8,9,10,11,12,13,14,15);
}

// DPP16 butterfly reductions within each 16-lane row (wave32 -> two rows),
// using VOP2+DPP fused ops: dst = op(dpp(src0), src1), one instruction/step.
__device__ __forceinline__ float rowmax16(float x) {
    asm("v_max_num_f32_dpp %0, %0, %0 quad_perm:[1,0,3,2] row_mask:0xf bank_mask:0xf bound_ctrl:1\n\t"
        "v_max_num_f32_dpp %0, %0, %0 quad_perm:[2,3,0,1] row_mask:0xf bank_mask:0xf bound_ctrl:1\n\t"
        "v_max_num_f32_dpp %0, %0, %0 row_half_mirror row_mask:0xf bank_mask:0xf bound_ctrl:1\n\t"
        "v_max_num_f32_dpp %0, %0, %0 row_mirror row_mask:0xf bank_mask:0xf bound_ctrl:1"
        : "+v"(x));
    return x;
}
__device__ __forceinline__ float rowsum16(float x) {
    asm("v_add_f32_dpp %0, %0, %0 quad_perm:[1,0,3,2] row_mask:0xf bank_mask:0xf bound_ctrl:1\n\t"
        "v_add_f32_dpp %0, %0, %0 quad_perm:[2,3,0,1] row_mask:0xf bank_mask:0xf bound_ctrl:1\n\t"
        "v_add_f32_dpp %0, %0, %0 row_half_mirror row_mask:0xf bank_mask:0xf bound_ctrl:1\n\t"
        "v_add_f32_dpp %0, %0, %0 row_mirror row_mask:0xf bank_mask:0xf bound_ctrl:1"
        : "+v"(x));
    return x;
}

__device__ __forceinline__ float elu1(float x) {
    return (x > 0.f) ? (x + 1.f) : __expf(x);
}

// pk interleave within each 32-key chunk: key j -> slot
//   32*(j>>5) + 2*(j&15) + ((j>>4)&1)
// Contiguous slots [32c+16*hi, +16) are exactly the A-layout K set for that
// half-wave and k=32 chunk c.
__device__ __forceinline__ int pkslot(int j) {
    return 32 * (j >> 5) + 2 * (j & 15) + ((j >> 4) & 1);
}

// ---------------------------------------------------------------------------
// Kernel 0: RoPE tables cos/sin[pos][col], pos<1024, col<64.
// ---------------------------------------------------------------------------
__global__ __launch_bounds__(256)
void ropetab_kernel(float* __restrict__ ctab, float* __restrict__ stab)
{
    int idx = blockIdx.x * 256 + threadIdx.x;   // 65536
    int pos = idx >> 6, col = idx & 63;
    int j = col & 31;
    float inv = __expf(-0.28782313662425575f * (float)j); // ln(10000)/32
    float ang = (float)pos * inv;
    float s, c;
    sincosf(ang, &s, &c);
    ctab[idx] = c;
    stab[idx] = s;
}

// ---------------------------------------------------------------------------
// Kernel 1: per (b,h,seg) segment summary: Kmat = sigma_k^T @ v (64x64), nvec.
// ---------------------------------------------------------------------------
__global__ __launch_bounds__(256)
void segsum_kernel(const float* __restrict__ k, const float* __restrict__ v,
                   float* __restrict__ Kmat, float* __restrict__ nvec)
{
    __shared__ alignas(16) float kch[32][64];
    __shared__ alignas(16) float vch[32][64];
    const int bhs = blockIdx.x;                 // bh*8 + seg
    const long base = (long)bhs * SEG * DD;
    const float* kp = k + base;
    const float* vp = v + base;
    const int tid = threadIdx.x;
    const int e  = tid >> 2;
    const int f0 = (tid & 3) << 4;

    float acc[16];
#pragma unroll
    for (int i = 0; i < 16; ++i) acc[i] = 0.f;
    float nacc = 0.f;

    for (int ch = 0; ch < 32; ++ch) {
        const int j0 = ch * 32;
        __syncthreads();
#pragma unroll
        for (int i = 0; i < 2; ++i) {
            int flat4 = tid + 256 * i;          // 512 quads
            int r = flat4 >> 4, cc = (flat4 & 15) << 2;
            f32x4 kx = *(const f32x4*)(kp + (j0 + r) * DD + cc);
            f32x4 vx = *(const f32x4*)(vp + (j0 + r) * DD + cc);
            f32x4 ke;
            ke.x = elu1(kx.x); ke.y = elu1(kx.y); ke.z = elu1(kx.z); ke.w = elu1(kx.w);
            *(f32x4*)&kch[r][cc] = ke;
            *(f32x4*)&vch[r][cc] = vx;
        }
        __syncthreads();
        for (int j = 0; j < 32; ++j) {
            float ke = kch[j][e];
#pragma unroll
            for (int i = 0; i < 16; ++i) acc[i] += ke * vch[j][f0 + i];
        }
        if (tid < 64) {
            for (int j = 0; j < 32; ++j) nacc += kch[j][tid];
        }
    }
    float* Ko = Kmat + (long)bhs * 4096;
#pragma unroll
    for (int i = 0; i < 4; ++i)
        *(f32x4*)&Ko[e * 64 + f0 + i * 4] =
            f32x4{acc[i*4+0], acc[i*4+1], acc[i*4+2], acc[i*4+3]};
    if (tid < 64) nvec[(long)bhs * 64 + tid] = nacc;
}

// ---------------------------------------------------------------------------
// Kernel 2: in-place exclusive prefix sum over segments per (b,h).
// ---------------------------------------------------------------------------
__global__ __launch_bounds__(256)
void prefix_kernel(float* __restrict__ Kmat, float* __restrict__ nvec)
{
    const int bh = blockIdx.x;
    const int tid = threadIdx.x;
    float carry[16];
#pragma unroll
    for (int i = 0; i < 16; ++i) carry[i] = 0.f;
    for (int s = 0; s < NSEG; ++s) {
        float* p = Kmat + (long)(bh * NSEG + s) * 4096 + tid * 16;
#pragma unroll
        for (int i = 0; i < 16; ++i) { float t = p[i]; p[i] = carry[i]; carry[i] += t; }
    }
    if (tid < 64) {
        float c = 0.f;
        for (int s = 0; s < NSEG; ++s) {
            float* p = nvec + (long)(bh * NSEG + s) * 64 + tid;
            float t = *p; *p = c; c += t;
        }
    }
}

// ---------------------------------------------------------------------------
// Kernel 3: mem_out = sigma_q @ M / (sigma_q . n + eps); writes out = g*mem_out
// ---------------------------------------------------------------------------
__global__ __launch_bounds__(256)
void memout_kernel(const float* __restrict__ q, const float* __restrict__ gate,
                   const float* __restrict__ Kmat, const float* __restrict__ nvec,
                   float* __restrict__ out)
{
    __shared__ alignas(16) float M[64 * 64];
    __shared__ alignas(16) float N[64];
    __shared__ alignas(16) float sq[64][64];
    const int bhs = blockIdx.x;
    const int hh = (bhs >> 3) & 7;
    const int tid = threadIdx.x;
    const long base = (long)bhs * SEG * DD;
    const float* qp = q + base;
    float* op = out + base;
    const float gg = 1.f / (1.f + __expf(-gate[hh]));

#pragma unroll
    for (int i = 0; i < 4; ++i) {
        int q4 = tid + 256 * i;                 // 1024 quads
        *(f32x4*)&M[q4 * 4] = *(const f32x4*)(Kmat + (long)bhs * 4096 + q4 * 4);
    }
    if (tid < 64) N[tid] = nvec[(long)bhs * 64 + tid];

    const int r  = tid >> 2;
    const int f0 = (tid & 3) << 4;
    for (int ch = 0; ch < 16; ++ch) {
        const int row0 = ch * 64;
        __syncthreads();
#pragma unroll
        for (int i = 0; i < 4; ++i) {
            int flat4 = tid + 256 * i;
            int rr = flat4 >> 4, cc = (flat4 & 15) << 2;
            f32x4 x = *(const f32x4*)(qp + (row0 + rr) * DD + cc);
            f32x4 sx;
            sx.x = elu1(x.x); sx.y = elu1(x.y); sx.z = elu1(x.z); sx.w = elu1(x.w);
            *(f32x4*)&sq[rr][cc] = sx;
        }
        __syncthreads();
        float acc[16];
#pragma unroll
        for (int i = 0; i < 16; ++i) acc[i] = 0.f;
        float nt = 0.f;
        for (int d = 0; d < 64; ++d) {
            float sv = sq[r][d];
            nt += sv * N[d];
            const float* Mr = M + d * 64 + f0;
#pragma unroll
            for (int i = 0; i < 16; ++i) acc[i] += sv * Mr[i];
        }
        const float invden = gg / (nt + 1e-6f);
        float* orow = op + (long)(row0 + r) * DD + f0;
#pragma unroll
        for (int i = 0; i < 4; ++i)
            *(f32x4*)&orow[i * 4] = f32x4{acc[i*4+0] * invden, acc[i*4+1] * invden,
                                          acc[i*4+2] * invden, acc[i*4+3] * invden};
    }
}

// ---------------------------------------------------------------------------
// Kernel 4: in-segment causal flash attention, f16 WMMA / f32 accumulate.
// grid = b*h*nseg*(1024/128) = 2048 blocks, 256 threads (8 waves).
// Wave w owns q rows [16w, 16w+16) of its 128-row tile; key blocks of 64.
// Wave-uniform (scalarized) skip of fully-masked key blocks.
// ---------------------------------------------------------------------------
__global__ __launch_bounds__(256)
void attn_kernel(const float* __restrict__ q, const float* __restrict__ k,
                 const float* __restrict__ v, const float* __restrict__ gate,
                 const float* __restrict__ ctab, const float* __restrict__ stab,
                 float* __restrict__ out)
{
    __shared__ alignas(32) _Float16 qbuf[QTILE][72];   // q tile, RoPE'd, *0.125
    __shared__ alignas(32) _Float16 kbuf[KTILE][72];   // k block, RoPE'd
    __shared__ alignas(32) _Float16 vtbuf[DD][72];     // v^T [d][pk-slot 0..63]
    __shared__ alignas(32) _Float16 pbuf[8][16][64];   // per-wave P [row][pk-slot]

    const int blk = blockIdx.x;
    const int qt  = blk & 7;                    // 8 q-tiles of 128 per segment
    const int seg = (blk >> 3) & 7;
    const int bh  = blk >> 6;
    const int hh  = bh & 7;
    const long base = ((long)bh * SS + (long)seg * SEG) * DD;
    const float* qp = q + base;
    const float* kp = k + base;
    const float* vp = v + base;
    float* op = out + base;

    const float gg  = 1.f / (1.f + __expf(-gate[hh]));
    const float wat = 1.f - gg;

    const int tid  = threadIdx.x;
    const int w    = tid >> 5;                  // 0..7
    const int lane = tid & 31;
    const int hi   = lane >> 4;
    const int l2   = lane & 15;
    const int qbase = qt * QTILE;

    // Stage Q tile: table RoPE + 1/sqrt(64). 2048 quads, b128 loads.
#pragma unroll
    for (int i = 0; i < 8; ++i) {
        int flat4 = tid + 256 * i;
        int r = flat4 >> 4, c0 = (flat4 & 15) << 2;
        const f32x4 x  = *(const f32x4*)(qp + (qbase + r) * DD + c0);
        const f32x4 xr = *(const f32x4*)(qp + (qbase + r) * DD + (c0 ^ 32));
        const f32x4 cs = *(const f32x4*)(ctab + (qbase + r) * DD + c0);
        const f32x4 sn = *(const f32x4*)(stab + (qbase + r) * DD + c0);
        const float sg = (c0 < 32) ? -1.f : 1.f;
        f32x4 val = (x * cs + (sg * xr) * sn) * 0.125f;
        *(v4h*)&qbuf[r][c0] = __builtin_convertvector(val, v4h);
    }
    __syncthreads();

    // A operand for Q (two k=32 chunks over d=64), per WMMA A layout.
    const int arow = 16 * w + l2;
    v16h qa[2];
#pragma unroll
    for (int c = 0; c < 2; ++c) {
        const v8h* p0 = (const v8h*)&qbuf[arow][c * 32 + 8 * hi];
        const v8h* p1 = (const v8h*)&qbuf[arow][c * 32 + 16 + 8 * hi];
        qa[c] = cat8(*p0, *p1);
    }

    const v8f zero = {0.f,0.f,0.f,0.f,0.f,0.f,0.f,0.f};
    v8f o[4] = {zero, zero, zero, zero};
    float mi[8], li[8];
#pragma unroll
    for (int r = 0; r < 8; ++r) { mi[r] = -1e30f; li[r] = 0.f; }

    const int wmax = qbase + 16 * w + 15;       // last row this wave owns
    const int nkb = 2 * (qt + 1);               // 64-key blocks for causality
    for (int kb = 0; kb < nkb; ++kb) {
        const int kb0 = kb * KTILE;
        // wave-uniform skip condition, forced scalar (keeps EXEC untouched)
        const int active = __builtin_amdgcn_readfirstlane((kb0 <= wmax) ? 1 : 0);
        __syncthreads();
        // Stage K block (table RoPE) row-major [key][d]. 1024 quads.
#pragma unroll
        for (int i = 0; i < 4; ++i) {
            int flat4 = tid + 256 * i;
            int r = flat4 >> 4, c0 = (flat4 & 15) << 2;
            const f32x4 x  = *(const f32x4*)(kp + (kb0 + r) * DD + c0);
            const f32x4 xr = *(const f32x4*)(kp + (kb0 + r) * DD + (c0 ^ 32));
            const f32x4 cs = *(const f32x4*)(ctab + (kb0 + r) * DD + c0);
            const f32x4 sn = *(const f32x4*)(stab + (kb0 + r) * DD + c0);
            const float sg = (c0 < 32) ? -1.f : 1.f;
            f32x4 val = x * cs + (sg * xr) * sn;
            *(v4h*)&kbuf[r][c0] = __builtin_convertvector(val, v4h);
        }
        // Stage V transposed with interleaved key slot. 1024 quads.
#pragma unroll
        for (int i = 0; i < 4; ++i) {
            int flat4 = tid + 256 * i;
            int r = flat4 >> 4, c0 = (flat4 & 15) << 2;   // r = key
            f32x4 x = *(const f32x4*)(vp + (kb0 + r) * DD + c0);
            v4h h = __builtin_convertvector(x, v4h);
            const int ps = pkslot(r);
            vtbuf[c0 + 0][ps] = h.x; vtbuf[c0 + 1][ps] = h.y;
            vtbuf[c0 + 2][ps] = h.z; vtbuf[c0 + 3][ps] = h.w;
        }
        __syncthreads();

        if (active) {
            // S = Q @ K^T : four 16x16 N-tiles, k=64 via 2 chained WMMAs each.
            v8f sacc[4];
#pragma unroll
            for (int nt = 0; nt < 4; ++nt) {
                v8f acc = zero;
                const int keyrow = nt * 16 + l2;    // B layout: N = lane%16
#pragma unroll
                for (int c = 0; c < 2; ++c) {
                    const v8h* b0 = (const v8h*)&kbuf[keyrow][c * 32 + 8 * hi];
                    const v8h* b1 = (const v8h*)&kbuf[keyrow][c * 32 + 16 + 8 * hi];
                    v16h bk = cat8(*b0, *b1);
                    acc = __builtin_amdgcn_wmma_f32_16x16x32_f16(
                            false, qa[c], false, bk, (short)0, acc, false, false);
                }
                sacc[nt] = acc;
            }

            // Online softmax (C-layout: elem r -> row r+8*hi, col l2).
#pragma unroll
            for (int r = 0; r < 8; ++r) {
                const int qrow = qbase + 16 * w + 8 * hi + r;
                float s[4];
#pragma unroll
                for (int nt = 0; nt < 4; ++nt) {
                    s[nt] = sacc[nt][r];
                    if (kb0 + nt * 16 + l2 > qrow) s[nt] = -1e30f;
                }
                float mx = fmaxf(fmaxf(s[0], s[1]), fmaxf(s[2], s[3]));
                mx = rowmax16(mx);
                const float mnew  = fmaxf(mi[r], mx);
                const float alpha = __expf(mi[r] - mnew);
                float p[4];
#pragma unroll
                for (int nt = 0; nt < 4; ++nt) p[nt] = __expf(s[nt] - mnew);
                float rs = (p[0] + p[1]) + (p[2] + p[3]);
                rs = rowsum16(rs);
                li[r] = li[r] * alpha + rs;
                mi[r] = mnew;
#pragma unroll
                for (int t = 0; t < 4; ++t) o[t][r] *= alpha;
                v2h pr0; pr0.x = (_Float16)p[0]; pr0.y = (_Float16)p[1];
                v2h pr1; pr1.x = (_Float16)p[2]; pr1.y = (_Float16)p[3];
                *(v2h*)&pbuf[w][8 * hi + r][2 * l2]      = pr0; // keys l2, l2+16
                *(v2h*)&pbuf[w][8 * hi + r][32 + 2 * l2] = pr1; // keys l2+32, l2+48
            }
        }
        __syncthreads();

        if (active) {
            // O += P @ V over two k=32 chunks. A = P: contiguous 16-half load
            // (interleaved slots match A-layout K set per half-wave & chunk).
#pragma unroll
            for (int c = 0; c < 2; ++c) {
                const v16h pa = *(const v16h*)&pbuf[w][l2][c * 32 + 16 * hi];
#pragma unroll
                for (int t = 0; t < 4; ++t) {
                    const int col = t * 16 + l2;
                    const v8h* v0 = (const v8h*)&vtbuf[col][c * 32 + 16 * hi];
                    const v8h* v1 = (const v8h*)&vtbuf[col][c * 32 + 16 * hi + 8];
                    v16h vb = cat8(*v0, *v1);
                    o[t] = __builtin_amdgcn_wmma_f32_16x16x32_f16(
                             false, pa, false, vb, (short)0, o[t], false, false);
                }
            }
        }
    }

    // Finalize: out += (1-g) * O / l  (out already holds g*mem_out)
#pragma unroll
    for (int r = 0; r < 8; ++r) {
        const float sc = wat / li[r];
        const int row = qbase + 16 * w + 8 * hi + r;
#pragma unroll
        for (int t = 0; t < 4; ++t) {
            const int col = t * 16 + l2;
            float* dst = op + (long)row * DD + col;
            *dst = *dst + o[t][r] * sc;
        }
    }
}

// ---------------------------------------------------------------------------
extern "C" void kernel_launch(void* const* d_in, const int* in_sizes, int n_in,
                              void* d_out, int out_size, void* d_ws, size_t ws_size,
                              hipStream_t stream) {
    (void)in_sizes; (void)n_in; (void)out_size; (void)ws_size;
    const float* q    = (const float*)d_in[0];
    const float* k    = (const float*)d_in[1];
    const float* v    = (const float*)d_in[2];
    const float* gate = (const float*)d_in[3];
    float* out  = (float*)d_out;
    float* Kmat = (float*)d_ws;                      // 256*4096 floats
    float* nvec = Kmat + (size_t)256 * 4096;         // 256*64 floats
    float* ctab = nvec + (size_t)256 * 64;           // 1024*64 floats
    float* stab = ctab + (size_t)1024 * 64;          // 1024*64 floats

    ropetab_kernel<<<256, 256, 0, stream>>>(ctab, stab);
    segsum_kernel<<<BB * HH * NSEG, 256, 0, stream>>>(k, v, Kmat, nvec);
    prefix_kernel<<<BB * HH, 256, 0, stream>>>(Kmat, nvec);
    memout_kernel<<<BB * HH * NSEG, 256, 0, stream>>>(q, gate, Kmat, nvec, out);
    attn_kernel<<<BB * HH * NSEG * (SEG / QTILE), 256, 0, stream>>>(
        q, k, v, gate, ctab, stab, out);
}
